// SelfAttn_64888365908362
// MI455X (gfx1250) — matrix operand order repaired and checked by reading, not compiled, not run on hardware
//
#include <hip/hip_runtime.h>

// MI455X (gfx1250, wave32) fused self-attention.
// bf16 WMMA (v_wmma_f32_16x16x32_bf16) everywhere; flash-style attention so the
// 4096x4096 score matrix never touches HBM (would be 512MB/batch vs 23.3TB/s).

typedef __attribute__((ext_vector_type(16))) __bf16 v16bf;
typedef __attribute__((ext_vector_type(8)))  __bf16 v8bf;
typedef __attribute__((ext_vector_type(8)))  float  v8f;

#define NPOS 4096
#define CIN  512
#define KF   64
#define CHD  256
#define MALL 384
#define NB   8

static __device__ __forceinline__ v16bf mk16(v8bf lo, v8bf hi) {
  v16bf r;
#pragma unroll
  for (int i = 0; i < 8; ++i) { r[i] = lo[i]; r[i + 8] = hi[i]; }
  return r;
}
// A fragment (16x32 MxK): lane half h, row m=lane&15.
// a[i] = A[m][k0 + h*8 + i]      for i<8
// a[i] = A[m][k0 + h*8 + 16 + i-8] for i>=8   (per ISA 16-bit A layout)
static __device__ __forceinline__ v16bf load_afrag(const __bf16* row, int k0, int half) {
  const v8bf* p0 = (const v8bf*)(row + k0 + half * 8);
  const v8bf* p1 = (const v8bf*)(row + k0 + half * 8 + 16);
  return mk16(*p0, *p1);
}
// B fragment (32x16 KxN) read from LDS stored as [n][k]:
// b[i] = B[k0 + half*8 + i][n=lane&15] for i<8, B[k0 + half*8 + 16 + i-8][n] for i>=8.
static __device__ __forceinline__ v16bf load_bfrag(const __bf16* row, int k0, int half) {
  const v8bf* p0 = (const v8bf*)(row + k0 + half * 8);
  const v8bf* p1 = (const v8bf*)(row + k0 + half * 8 + 16);
  return mk16(*p0, *p1);
}
static __device__ __forceinline__ v8f wmma_bf16(v16bf a, v16bf b, v8f c) {
  return __builtin_amdgcn_wmma_f32_16x16x32_bf16(false, a, false, b, (short)0, c,
                                                 false, false);
}

// ---------------------------------------------------------------- pack weights
__global__ __launch_bounds__(256) void pack_kernel(
    const float* __restrict__ Wf, const float* __restrict__ Wg,
    const float* __restrict__ Wh, const float* __restrict__ Wv,
    const float* __restrict__ bf, const float* __restrict__ bg,
    const float* __restrict__ bh, __bf16* __restrict__ Wall,
    __bf16* __restrict__ Wvb, float* __restrict__ bias) {
  const int idx = blockIdx.x * 256 + threadIdx.x;
  if (idx < MALL * CIN) {
    const int m = idx >> 9, k = idx & (CIN - 1);
    float v = (m < 64)    ? Wf[m * CIN + k]
              : (m < 128) ? Wg[(m - 64) * CIN + k]
                          : Wh[(m - 128) * CIN + k];
    Wall[idx] = (__bf16)v;
  }
  if (idx < CIN * CHD) Wvb[idx] = (__bf16)Wv[idx];
  if (idx < MALL)
    bias[idx] = (idx < 64) ? bf[idx] : (idx < 128) ? bg[idx - 64] : bh[idx - 128];
}

// ------------------------------------------------- fgh = W_all(384x512) @ x + b
__global__ __launch_bounds__(256) void proj_kernel(
    const float* __restrict__ x, const __bf16* __restrict__ Wall,
    const float* __restrict__ bias, __bf16* __restrict__ fgh) {
  const int b  = blockIdx.z;
  const int mt = blockIdx.y * 64;
  const int nt = blockIdx.x * 64;
  const float* xb  = x + (size_t)b * CIN * NPOS;
  __bf16*     outb = fgh + (size_t)b * MALL * NPOS;
  __shared__ __attribute__((aligned(16))) __bf16 Bt[64][32];  // [pos][ch]
  const int tid = threadIdx.x;
  const int lane = tid & 31, wave = tid >> 5;
  const int half = lane >> 4, ln = lane & 15;
  const int msub  = wave >> 1;
  const int nsub0 = (wave & 1) * 2;
  const int cload = tid >> 3;        // 0..31 ch within K-chunk
  const int pload = (tid & 7) * 8;   // 0..56 pos
  v8f acc0 = {}, acc1 = {};
  for (int k0 = 0; k0 < CIN; k0 += 32) {
#pragma unroll
    for (int j = 0; j < 8; ++j)
      Bt[pload + j][cload] =
          (__bf16)xb[(size_t)(k0 + cload) * NPOS + nt + pload + j];
    __syncthreads();
    v16bf a  = load_afrag(Wall + (size_t)(mt + msub * 16 + ln) * CIN, k0, half);
    v16bf b0 = load_bfrag(&Bt[nsub0 * 16 + ln][0], 0, half);
    v16bf b1 = load_bfrag(&Bt[(nsub0 + 1) * 16 + ln][0], 0, half);
    acc0 = wmma_bf16(a, b0, acc0);
    acc1 = wmma_bf16(a, b1, acc1);
    __syncthreads();
  }
#pragma unroll
  for (int r = 0; r < 8; ++r) {
    const int   m  = mt + msub * 16 + r + half * 8;
    const float bi = bias[m];
    outb[(size_t)m * NPOS + nt + nsub0 * 16 + ln]       = (__bf16)(acc0[r] + bi);
    outb[(size_t)m * NPOS + nt + (nsub0 + 1) * 16 + ln] = (__bf16)(acc1[r] + bi);
  }
}

// -------------------------------------------------- flash attention core
// block: 64 queries, 8 waves = 4 query-groups x 2 channel-halves.
__global__ __launch_bounds__(256) void attn_kernel(const __bf16* __restrict__ fgh,
                                                   __bf16* __restrict__ O2) {
  const int b  = blockIdx.y;
  const int q0 = blockIdx.x * 64;
  const __bf16* fb = fgh + (size_t)b * MALL * NPOS;
  const __bf16* gb = fb + (size_t)KF * NPOS;
  const __bf16* hb = fb + (size_t)(2 * KF) * NPOS;
  __bf16* Ob = O2 + (size_t)b * NPOS * CHD;   // [pos][ch]

  __shared__ __attribute__((aligned(16))) __bf16 ft[64][64];      // [q][ch]
  __shared__ __attribute__((aligned(16))) __bf16 gt[64][64];      // [key][ch]
  __shared__ __attribute__((aligned(16))) __bf16 hs[256][64];     // [ch][key]
  __shared__ __attribute__((aligned(16))) __bf16 ps[8][16][64];   // per-wave P

  const int tid = threadIdx.x;
  const int lane = tid & 31, wave = tid >> 5;
  const int half = lane >> 4, ln = lane & 15;
  const int qg = wave >> 1, chh = wave & 1;

  {  // stage f^T tile: ft[q][ch] = f[ch][q0+q]
    const int ch = tid >> 2, qs = (tid & 3) * 16;
    const v8bf* s = (const v8bf*)(fb + (size_t)ch * NPOS + q0 + qs);
    v8bf d0 = s[0], d1 = s[1];
#pragma unroll
    for (int j = 0; j < 8; ++j) {
      ft[qs + j][ch]     = d0[j];
      ft[qs + 8 + j][ch] = d1[j];
    }
  }
  __syncthreads();
  v16bf af0 = load_afrag(&ft[qg * 16 + ln][0], 0, half);
  v16bf af1 = load_afrag(&ft[qg * 16 + ln][0], 32, half);

  float mrow[8], lrow[8];
#pragma unroll
  for (int r = 0; r < 8; ++r) { mrow[r] = -1e30f; lrow[r] = 0.f; }
  v8f zero = {};
  v8f o[8];
#pragma unroll
  for (int ct = 0; ct < 8; ++ct) o[ct] = zero;

  for (int j0 = 0; j0 < NPOS; j0 += 64) {
    {  // stage g^T tile (transposed) and hv tile (straight)
      const int ch = tid >> 2, ks = (tid & 3) * 16;
      const v8bf* s = (const v8bf*)(gb + (size_t)ch * NPOS + j0 + ks);
      v8bf d0 = s[0], d1 = s[1];
#pragma unroll
      for (int j = 0; j < 8; ++j) {
        gt[ks + j][ch]     = d0[j];
        gt[ks + 8 + j][ch] = d1[j];
      }
      const uint4* sh = (const uint4*)(hb + (size_t)tid * NPOS + j0);
      uint4*       dh = (uint4*)(&hs[tid][0]);
#pragma unroll
      for (int j = 0; j < 8; ++j) dh[j] = sh[j];
      if (j0 + 64 < NPOS) {
        __builtin_prefetch(gb + (size_t)ch * NPOS + j0 + 64 + ks, 0, 0);
        __builtin_prefetch(hb + (size_t)tid * NPOS + j0 + 64, 0, 0);
      }
    }
    __syncthreads();

    // S = f^T g  : 4 tiles of 16x16, K=64 -> 2 WMMAs each
    v8f st[4] = {};
#pragma unroll
    for (int t = 0; t < 4; ++t) {
      v16bf b0 = load_bfrag(&gt[t * 16 + ln][0], 0, half);
      v16bf b1 = load_bfrag(&gt[t * 16 + ln][0], 32, half);
      st[t] = wmma_bf16(af0, b0, st[t]);
      st[t] = wmma_bf16(af1, b1, st[t]);
    }

    // online softmax: rows live at (VGPR r, lane-half); reduce over 16-lane groups
    float mnew8[8], sc8[8];
#pragma unroll
    for (int r = 0; r < 8; ++r) {
      float v = fmaxf(fmaxf(st[0][r], st[1][r]), fmaxf(st[2][r], st[3][r]));
      v = fmaxf(v, __shfl_xor(v, 1, 16));
      v = fmaxf(v, __shfl_xor(v, 2, 16));
      v = fmaxf(v, __shfl_xor(v, 4, 16));
      v = fmaxf(v, __shfl_xor(v, 8, 16));
      float mnew = fmaxf(mrow[r], v);
      sc8[r]   = __expf(mrow[r] - mnew);
      mrow[r]  = mnew;
      mnew8[r] = mnew;
    }
    float psum[8];
#pragma unroll
    for (int r = 0; r < 8; ++r) psum[r] = 0.f;
#pragma unroll
    for (int t = 0; t < 4; ++t) {
#pragma unroll
      for (int r = 0; r < 8; ++r) {
        float p = __expf(st[t][r] - mnew8[r]);
        psum[r] += p;
        ps[wave][r + half * 8][t * 16 + ln] = (__bf16)p;  // C-layout -> LDS
      }
    }
#pragma unroll
    for (int r = 0; r < 8; ++r) {
      float s = psum[r];
      s += __shfl_xor(s, 1, 16);
      s += __shfl_xor(s, 2, 16);
      s += __shfl_xor(s, 4, 16);
      s += __shfl_xor(s, 8, 16);
      lrow[r] = lrow[r] * sc8[r] + s;
    }
#pragma unroll
    for (int ct = 0; ct < 8; ++ct) {
#pragma unroll
      for (int r = 0; r < 8; ++r) o[ct][r] *= sc8[r];
    }
    asm volatile("" ::: "memory");  // order ps writes before A-frag reads (same wave)

    // O += P * hv^T : per-wave 16q x 128ch (8 tiles), K=64 keys
    v16bf ap0 = load_afrag(&ps[wave][ln][0], 0, half);
    v16bf ap1 = load_afrag(&ps[wave][ln][0], 32, half);
#pragma unroll
    for (int ct = 0; ct < 8; ++ct) {
      const __bf16* rh = &hs[chh * 128 + ct * 16 + ln][0];
      v16bf b0 = load_bfrag(rh, 0, half);
      v16bf b1 = load_bfrag(rh, 32, half);
      o[ct] = wmma_bf16(ap0, b0, o[ct]);
      o[ct] = wmma_bf16(ap1, b1, o[ct]);
    }
    __syncthreads();
  }

  float inv8[8];
#pragma unroll
  for (int r = 0; r < 8; ++r) inv8[r] = 1.0f / lrow[r];
#pragma unroll
  for (int ct = 0; ct < 8; ++ct) {
#pragma unroll
    for (int r = 0; r < 8; ++r) {
      const int q = q0 + qg * 16 + r + half * 8;
      const int c = chh * 128 + ct * 16 + ln;
      Ob[(size_t)q * CHD + c] = (__bf16)(o[ct][r] * inv8[r]);
    }
  }
}

// -------------------------------- out = gamma*(Wv(512x256) @ O + bv) + x
__global__ __launch_bounds__(256) void outproj_kernel(
    const float* __restrict__ x, const __bf16* __restrict__ Wvb,
    const float* __restrict__ bv, const float* __restrict__ gamma,
    const __bf16* __restrict__ O2, float* __restrict__ out) {
  const int b  = blockIdx.z;
  const int mt = blockIdx.y * 64;
  const int nt = blockIdx.x * 64;
  const __bf16* Ob = O2 + (size_t)b * NPOS * CHD;
  __shared__ __attribute__((aligned(16))) __bf16 Bt[64][32];  // [pos][ch]
  const int tid = threadIdx.x;
  const int lane = tid & 31, wave = tid >> 5;
  const int half = lane >> 4, ln = lane & 15;
  const int msub  = wave >> 1;
  const int nsub0 = (wave & 1) * 2;
  const int pl = tid >> 2;        // 0..63
  const int cl = (tid & 3) * 8;   // 0,8,16,24
  v8f acc0 = {}, acc1 = {};
  for (int k0 = 0; k0 < CHD; k0 += 32) {
    *(v8bf*)(&Bt[pl][cl]) =
        *(const v8bf*)(Ob + (size_t)(nt + pl) * CHD + k0 + cl);
    __syncthreads();
    v16bf a  = load_afrag(Wvb + (size_t)(mt + msub * 16 + ln) * CHD, k0, half);
    v16bf b0 = load_bfrag(&Bt[nsub0 * 16 + ln][0], 0, half);
    v16bf b1 = load_bfrag(&Bt[(nsub0 + 1) * 16 + ln][0], 0, half);
    acc0 = wmma_bf16(a, b0, acc0);
    acc1 = wmma_bf16(a, b1, acc1);
    __syncthreads();
  }
  const float ga = gamma[0];
#pragma unroll
  for (int r = 0; r < 8; ++r) {
    const int    m    = mt + msub * 16 + r + half * 8;
    const float  bi   = bv[m];
    const size_t base = ((size_t)b * CIN + m) * NPOS + nt;
    const int p0 = nsub0 * 16 + ln, p1 = (nsub0 + 1) * 16 + ln;
    out[base + p0] = ga * (acc0[r] + bi) + x[base + p0];
    out[base + p1] = ga * (acc1[r] + bi) + x[base + p1];
  }
}

extern "C" void kernel_launch(void* const* d_in, const int* in_sizes, int n_in,
                              void* d_out, int out_size, void* d_ws, size_t ws_size,
                              hipStream_t stream) {
  (void)in_sizes; (void)n_in; (void)out_size; (void)ws_size;
  const float* x     = (const float*)d_in[0];
  const float* Wf    = (const float*)d_in[1];
  const float* bf    = (const float*)d_in[2];
  const float* Wg    = (const float*)d_in[3];
  const float* bg    = (const float*)d_in[4];
  const float* Wh    = (const float*)d_in[5];
  const float* bh    = (const float*)d_in[6];
  const float* Wv    = (const float*)d_in[7];
  const float* bv    = (const float*)d_in[8];
  const float* gamma = (const float*)d_in[9];
  float* out = (float*)d_out;

  char* ws = (char*)d_ws;
  constexpr size_t OFF_WALL = 0;
  constexpr size_t OFF_WV   = OFF_WALL + (size_t)MALL * CIN * 2;
  constexpr size_t OFF_BIAS = OFF_WV + (size_t)CIN * CHD * 2;
  constexpr size_t OFF_FGH  = (OFF_BIAS + (size_t)MALL * 4 + 255) & ~(size_t)255;
  constexpr size_t OFF_O2   = OFF_FGH + (size_t)NB * MALL * NPOS * 2;
  __bf16* Wall = (__bf16*)(ws + OFF_WALL);
  __bf16* Wvb  = (__bf16*)(ws + OFF_WV);
  float*  bias = (float*)(ws + OFF_BIAS);
  __bf16* fgh  = (__bf16*)(ws + OFF_FGH);
  __bf16* O2   = (__bf16*)(ws + OFF_O2);

  hipLaunchKernelGGL(pack_kernel, dim3((MALL * CIN + 255) / 256), dim3(256), 0,
                     stream, Wf, Wg, Wh, Wv, bf, bg, bh, Wall, Wvb, bias);
  hipLaunchKernelGGL(proj_kernel, dim3(NPOS / 64, MALL / 64, NB), dim3(256), 0,
                     stream, x, Wall, bias, fgh);
  hipLaunchKernelGGL(attn_kernel, dim3(NPOS / 64, NB), dim3(256), 0, stream, fgh, O2);
  hipLaunchKernelGGL(outproj_kernel, dim3(NPOS / 64, CIN / 64, NB), dim3(256), 0,
                     stream, x, Wvb, bv, gamma, O2, out);
}
